// SelfReg_32736240730145
// MI455X (gfx1250) — matrix operand order repaired and checked by reading, not compiled, not run on hardware
//
#include <hip/hip_runtime.h>

typedef unsigned short u16;
typedef unsigned int   u32;
typedef unsigned long long u64;

typedef __attribute__((ext_vector_type(16))) __bf16 v16bf;
typedef __attribute__((ext_vector_type(8)))  float  v8f;
typedef __attribute__((ext_vector_type(4)))  unsigned int u32x4;
typedef __attribute__((ext_vector_type(8)))  int          i32x8;
typedef __attribute__((ext_vector_type(4)))  int          i32x4;

#define B_ROWS 8192
#define DIM    2048
#define NCLS   345
#define LDO    384      // 345 padded to 3*128
#define EPSV   1e-5f

#define BM 128
#define BN 128
#define BK 32

__device__ __forceinline__ u16 f32_to_bf16(float f) {
  u32 u = __float_as_uint(f);
  u32 r = u + 0x7FFFu + ((u >> 16) & 1u);
  return (u16)(r >> 16);
}

__device__ __forceinline__ u32 pk_lo(u32 a, u32 b) { return (a & 0xFFFFu) | (b << 16); }
__device__ __forceinline__ u32 pk_hi(u32 a, u32 b) { return (a >> 16) | (b & 0xFFFF0000u); }

// ---------------------------------------------------------------------------
// Tensor Data Mover: 2D tile (global -> LDS), data_size = 8 bytes.
// tile_d0u: tile width in 8B units; tile_d1: rows; strides/dims in 8B units.
// ---------------------------------------------------------------------------
__device__ __forceinline__ void tdm_load_2d(u32 lds_off, const void* gptr,
                                            u32 tile_d0u, u32 tile_d1,
                                            u32 stride0u, u32 tensor_d0u, u32 tensor_d1) {
  u64 ga = (u64)gptr;
  u32x4 g0;
  g0[0] = 1u;                                             // count=1, user descriptor
  g0[1] = lds_off;                                        // LDS byte address
  g0[2] = (u32)(ga & 0xFFFFFFFFu);                        // global_addr[31:0]
  g0[3] = (u32)((ga >> 32) & 0x01FFFFFFu) | (2u << 30);   // global_addr[56:32] | type=2
  i32x8 g1;
  g1[0] = (int)(3u << 16);                                // data_size = 8B, no mask/flags
  g1[1] = (int)((tensor_d0u & 0xFFFFu) << 16);            // tensor_dim0[15:0]
  g1[2] = (int)((tensor_d0u >> 16) | ((tensor_d1 & 0xFFFFu) << 16));
  g1[3] = (int)((tensor_d1 >> 16) | (tile_d0u << 16));    // tile_dim0
  g1[4] = (int)(tile_d1 & 0xFFFFu);                       // tile_dim1 (tile_dim2=0)
  g1[5] = (int)stride0u;                                  // tensor_dim0_stride[31:0]
  g1[6] = 0;                                              // stride0[47:32] | stride1[15:0]
  g1[7] = 0;                                              // stride1[47:16]
  i32x4 z4 = {0, 0, 0, 0};
#if __clang_major__ >= 23
  i32x8 z8 = {0, 0, 0, 0, 0, 0, 0, 0};
  __builtin_amdgcn_tensor_load_to_lds(g0, g1, z4, z4, z8, 0);
#else
  __builtin_amdgcn_tensor_load_to_lds(g0, g1, z4, z4, 0);
#endif
}

__device__ __forceinline__ void pack_store_B(uint4 r0, uint4 r1, u32* dst) {
  uint4 q0, q1;
  q0.x = pk_lo(r0.x, r1.x); q0.y = pk_hi(r0.x, r1.x);
  q0.z = pk_lo(r0.y, r1.y); q0.w = pk_hi(r0.y, r1.y);
  q1.x = pk_lo(r0.z, r1.z); q1.y = pk_hi(r0.z, r1.z);
  q1.z = pk_lo(r0.w, r1.w); q1.w = pk_hi(r0.w, r1.w);
  uint4* sp = reinterpret_cast<uint4*>(dst);
  sp[0] = q0; sp[1] = q1;
}

// ---------------------------------------------------------------------------
// Tiled bf16 WMMA GEMM:  C[M x N] = A[M x K] * B[K x N] + bias[N]
// Block = 256 threads (8 waves), tile 128x128, K step 32, double-buffered LDS.
// A tile staged by the Tensor Data Mover (wave 0), B tile K-pair packed by all
// waves. Wave (wrow, wcol) owns a 64x32 region = 4x2 WMMA tiles.
// ---------------------------------------------------------------------------
__global__ __launch_bounds__(256) void k_gemm_bf16(
    const u16* __restrict__ A, int lda,
    const u16* __restrict__ Bm, int ldb,
    const float* __restrict__ bias,
    float* __restrict__ C, int ldc, int K)
{
  __shared__ u16 sA[2][BM * BK];          // row-major [m][k], 8 KB each
  __shared__ u32 sB[2][(BK / 2) * BN];    // K-pair packed [j][n], 8 KB each

  const int tid   = threadIdx.x;
  const int lane  = tid & 31;
  const int wave  = tid >> 5;
  const int wrow  = wave & 1;
  const int wcol  = wave >> 1;
  const int lmod  = lane & 15;
  const int lhalf = (lane < 16) ? 0 : 8;
  const int jbase = (lane < 16) ? 0 : 4;
  const size_t bm0 = (size_t)blockIdx.y * BM;
  const size_t bn0 = (size_t)blockIdx.x * BN;

  const int bj  = tid >> 4;            // 0..15 (K pair index)
  const int bnb = (tid & 15) << 3;     // 0..120 (N base, 8 wide)
  const int nk  = K / BK;
  const u32 str0u = (u32)(lda >> 2);   // lda*2 bytes / 8

  v8f acc[4][2] = {};

  // ---------------- prologue: stage tile 0 into buffer 0 ----------------
  if (wave == 0)
    tdm_load_2d((u32)(u64)(void*)&sA[0][0], A + bm0 * (size_t)lda,
                (BK * 2) / 8, BM, str0u, str0u, (u32)BM);
  {
    uint4 r0 = *reinterpret_cast<const uint4*>(Bm + (size_t)(2 * bj) * (size_t)ldb + bn0 + bnb);
    uint4 r1 = *reinterpret_cast<const uint4*>(Bm + (size_t)(2 * bj + 1) * (size_t)ldb + bn0 + bnb);
    pack_store_B(r0, r1, &sB[0][bj * BN + bnb]);
  }
  if (wave == 0) __builtin_amdgcn_s_wait_tensorcnt(0);
  __syncthreads();

  for (int it = 0; it < nk; ++it) {
    const int cur = it & 1;
    const int nxt = cur ^ 1;
    const bool have_next = (it + 1) < nk;
    uint4 r0 = {}, r1 = {};
    if (have_next) {
      const int kn = (it + 1) * BK;
      if (wave == 0)
        tdm_load_2d((u32)(u64)(void*)&sA[nxt][0], A + bm0 * (size_t)lda + kn,
                    (BK * 2) / 8, BM, str0u, str0u, (u32)BM);
      r0 = *reinterpret_cast<const uint4*>(Bm + (size_t)(kn + 2 * bj) * (size_t)ldb + bn0 + bnb);
      r1 = *reinterpret_cast<const uint4*>(Bm + (size_t)(kn + 2 * bj + 1) * (size_t)ldb + bn0 + bnb);
      __builtin_prefetch(Bm + (size_t)(kn + BK + 2 * bj) * (size_t)ldb + bn0 + bnb, 0, 1);
    }

    // ---- fragments from current buffers ----
    union AFrag { uint4 q[2]; v16bf v; } af[4];
    union BFrag { u32 w[8];  v16bf v; } bfr[2];
#pragma unroll
    for (int mt = 0; mt < 4; ++mt) {
      const int row = wrow * 64 + mt * 16 + lmod;
      af[mt].q[0] = *reinterpret_cast<const uint4*>(&sA[cur][row * BK + lhalf]);
      af[mt].q[1] = *reinterpret_cast<const uint4*>(&sA[cur][row * BK + 16 + lhalf]);
    }
#pragma unroll
    for (int nt = 0; nt < 2; ++nt) {
      const int n = wcol * 32 + nt * 16 + lmod;
#pragma unroll
      for (int v = 0; v < 8; ++v) bfr[nt].w[v] = sB[cur][(jbase + v + (v & 4)) * BN + n];
    }
    // ---- 8 WMMAs (overlap with in-flight TDM + B loads) ----
#pragma unroll
    for (int mt = 0; mt < 4; ++mt)
#pragma unroll
      for (int nt = 0; nt < 2; ++nt)
        acc[mt][nt] = __builtin_amdgcn_wmma_f32_16x16x32_bf16(
            false, af[mt].v, false, bfr[nt].v, (short)0, acc[mt][nt], false, false);

    if (have_next)
      pack_store_B(r0, r1, &sB[nxt][bj * BN + bnb]);
    if (wave == 0) __builtin_amdgcn_s_wait_tensorcnt(0);
    __syncthreads();
  }

  // ---- epilogue: C/D layout: vgpr e -> M = e (+8 for lanes 16..31), N = lane%16
#pragma unroll
  for (int nt = 0; nt < 2; ++nt) {
    const size_t col = bn0 + wcol * 32 + nt * 16 + lmod;
    const float bv = bias[col];
#pragma unroll
    for (int mt = 0; mt < 4; ++mt) {
      const size_t rbase = bm0 + (size_t)(wrow * 64 + mt * 16 + ((lane < 16) ? 0 : 8));
#pragma unroll
      for (int e = 0; e < 8; ++e)
        C[(rbase + e) * (size_t)ldc + col] = acc[mt][nt][e] + bv;
    }
  }
}

// ---------------------------------------------------------------------------
// Conversions / gathers
// ---------------------------------------------------------------------------
__global__ void k_gather_bf16(const float* __restrict__ x, const int* __restrict__ order,
                              u16* __restrict__ dst) {
  size_t i = ((size_t)blockIdx.x * blockDim.x + threadIdx.x) * 4;
  if (i >= (size_t)B_ROWS * DIM) return;
  size_t r = i / DIM, c = i % DIM;
  const float* src = x + (size_t)order[r] * DIM + c;
#pragma unroll
  for (int m = 0; m < 4; ++m) dst[i + m] = f32_to_bf16(src[m]);
}

__global__ void k_conv_bf16(const float* __restrict__ s, u16* __restrict__ d, size_t n) {
  size_t i = ((size_t)blockIdx.x * blockDim.x + threadIdx.x) * 4;
  if (i >= n) return;
#pragma unroll
  for (int m = 0; m < 4; ++m) d[i + m] = f32_to_bf16(s[i + m]);
}

__global__ void k_pad_wc(const float* __restrict__ wc, const float* __restrict__ bc,
                         u16* __restrict__ wcp, float* __restrict__ bcp) {
  size_t i = (size_t)blockIdx.x * blockDim.x + threadIdx.x;
  const size_t tot = (size_t)DIM * NCLS;
  if (i < tot) {
    size_t k = i / NCLS, c = i % NCLS;
    wcp[k * LDO + c] = f32_to_bf16(wc[i]);
  }
  if (i < NCLS) bcp[i] = bc[i];
}

__global__ void k_relu_bf16(const float* __restrict__ z, u16* __restrict__ d, size_t n) {
  size_t i = ((size_t)blockIdx.x * blockDim.x + threadIdx.x) * 4;
  if (i >= n) return;
#pragma unroll
  for (int m = 0; m < 4; ++m) {
    float v = z[i + m];
    d[i + m] = f32_to_bf16(v > 0.f ? v : 0.f);
  }
}

// ---------------------------------------------------------------------------
// BatchNorm stats + apply
// ---------------------------------------------------------------------------
__global__ void k_colstats(const float* __restrict__ z, float* __restrict__ s,
                           float* __restrict__ ss, int rows) {
  int c  = blockIdx.x * blockDim.x + threadIdx.x;   // column
  int r0 = blockIdx.y * rows;
  float a = 0.f, b = 0.f;
  for (int r = r0; r < r0 + rows; ++r) {
    float v = z[(size_t)r * DIM + c];
    a += v; b += v * v;
  }
  atomicAdd(&s[c], a);
  atomicAdd(&ss[c], b);
}

// mode 0: relu -> bf16 out; mode 1: plain f32 out
__global__ void k_bn_apply(const float* __restrict__ z, const float* __restrict__ s,
                           const float* __restrict__ ss, const float* __restrict__ g,
                           const float* __restrict__ be, u16* __restrict__ obf,
                           float* __restrict__ of32, int mode) {
  size_t i = ((size_t)blockIdx.x * blockDim.x + threadIdx.x) * 4;
  if (i >= (size_t)B_ROWS * DIM) return;
  size_t c0 = i % DIM;
#pragma unroll
  for (int m = 0; m < 4; ++m) {
    size_t c = c0 + m;
    float mean = s[c] * (1.0f / B_ROWS);
    float var  = ss[c] * (1.0f / B_ROWS) - mean * mean;
    float v = (z[i + m] - mean) * rsqrtf(var + EPSV) * g[c] + be[c];
    if (mode == 0) {
      v = v > 0.f ? v : 0.f;
      obf[i + m] = f32_to_bf16(v);
    } else {
      of32[i + m] = v;
    }
  }
}

// ---------------------------------------------------------------------------
// Reductions / losses
// ---------------------------------------------------------------------------
__device__ __forceinline__ float blk_sum(float v) {
  __shared__ float sm[8];
#pragma unroll
  for (int o = 16; o > 0; o >>= 1) v += __shfl_xor(v, o, 32);
  int lane = threadIdx.x & 31, w = threadIdx.x >> 5;
  __syncthreads();
  if (lane == 0) sm[w] = v;
  __syncthreads();
  float r = (threadIdx.x < 8) ? sm[threadIdx.x] : 0.f;
#pragma unroll
  for (int o = 4; o > 0; o >>= 1) r += __shfl_xor(r, o, 32);
  return r;  // valid on thread 0
}

__device__ __forceinline__ float blk_max(float v) {
  __shared__ float sm[8];
#pragma unroll
  for (int o = 16; o > 0; o >>= 1) v = fmaxf(v, __shfl_xor(v, o, 32));
  int lane = threadIdx.x & 31, w = threadIdx.x >> 5;
  __syncthreads();
  if (lane == 0) sm[w] = v;
  __syncthreads();
  float r = (threadIdx.x < 8) ? sm[threadIdx.x] : -3.0e38f;
#pragma unroll
  for (int o = 4; o > 0; o >>= 1) r = fmaxf(r, __shfl_xor(r, o, 32));
  return r;  // valid on thread 0
}

// accs: [0]=ind_logit ss, [1]=hdl_logit ss, [2]=ind_feat ss, [3]=hdl_feat ss, [4]=ce sum
__global__ void k_loss_logits(const float* __restrict__ out, const int* __restrict__ s1,
                              const int* __restrict__ s2, const float* __restrict__ lamp,
                              float* __restrict__ accs) {
  int r = blockIdx.x;
  const float lam = lamp[0];
  const float* o  = out + (size_t)r * LDO;
  const float* o2 = out + (size_t)s1[r] * LDO;
  const float* o3 = out + (size_t)s2[r] * LDO;
  float aind = 0.f, ahdl = 0.f;
  for (int c = threadIdx.x; c < NCLS; c += blockDim.x) {
    float a = o[c], b = o2[c], d = o3[c];
    float di = a - b; aind += di * di;
    float mm = lam * b + (1.0f - lam) * d;
    float dh = a - mm; ahdl += dh * dh;
  }
  float t = blk_sum(aind);
  if (threadIdx.x == 0) atomicAdd(&accs[0], t);
  t = blk_sum(ahdl);
  if (threadIdx.x == 0) atomicAdd(&accs[1], t);
}

__global__ void k_loss_feat(const float* __restrict__ p, const int* __restrict__ s1,
                            const int* __restrict__ s2, const float* __restrict__ lamp,
                            float* __restrict__ accs) {
  int r = blockIdx.x;
  const float lam = lamp[0];
  const float* a = p + (size_t)r * DIM;
  const float* b = p + (size_t)s1[r] * DIM;
  const float* d = p + (size_t)s2[r] * DIM;
  float aind = 0.f, ahdl = 0.f;
  for (int c = threadIdx.x; c < DIM; c += blockDim.x) {
    float av = a[c], bv = b[c], dv = d[c];
    float di = av - bv; aind += di * di;
    float mm = lam * bv + (1.0f - lam) * dv;
    float dh = av - mm; ahdl += dh * dh;
  }
  float t = blk_sum(aind);
  if (threadIdx.x == 0) atomicAdd(&accs[2], t);
  t = blk_sum(ahdl);
  if (threadIdx.x == 0) atomicAdd(&accs[3], t);
}

__global__ void k_ce(const float* __restrict__ out, const int* __restrict__ y,
                     const int* __restrict__ order, float* __restrict__ accs) {
  int r = blockIdx.x;
  const float* o = out + (size_t)r * LDO;
  float mx = -3.0e38f;
  for (int c = threadIdx.x; c < NCLS; c += blockDim.x) mx = fmaxf(mx, o[c]);
  mx = blk_max(mx);
  __shared__ float smx;
  if (threadIdx.x == 0) smx = mx;
  __syncthreads();
  mx = smx;
  float se = 0.f;
  for (int c = threadIdx.x; c < NCLS; c += blockDim.x) se += __expf(o[c] - mx);
  se = blk_sum(se);
  if (threadIdx.x == 0) {
    int t = y[order[r]];
    atomicAdd(&accs[4], -(o[t] - mx - __logf(se)));
  }
}

__global__ void k_final(const float* __restrict__ accs, const float* __restrict__ lamp,
                        float* __restrict__ outp) {
  float lam = lamp[0];
  float Lil = accs[0] / ((float)B_ROWS * (float)NCLS);
  float Lhl = accs[1] / ((float)B_ROWS * (float)NCLS);
  float Lif = 0.3f * accs[2] / ((float)B_ROWS * (float)DIM);
  float Lhf = 0.3f * accs[3] / ((float)B_ROWS * (float)DIM);
  float cl  = accs[4] / (float)B_ROWS;
  float Cs  = fminf(cl, 1.0f);
  outp[0] = cl + Cs * (lam * (Lil + Lif) + (1.0f - lam) * (Lhl + Lhf));
}

// ---------------------------------------------------------------------------
extern "C" void kernel_launch(void* const* d_in, const int* in_sizes, int n_in,
                              void* d_out, int out_size, void* d_ws, size_t ws_size,
                              hipStream_t stream) {
  (void)in_sizes; (void)n_in; (void)out_size; (void)ws_size;
  const float* x   = (const float*)d_in[0];
  const float* lam = (const float*)d_in[1];
  const float* Wf  = (const float*)d_in[2];
  const float* bfv = (const float*)d_in[3];
  const float* W1  = (const float*)d_in[4];
  const float* b1  = (const float*)d_in[5];
  const float* g1  = (const float*)d_in[6];
  const float* be1 = (const float*)d_in[7];
  const float* W2  = (const float*)d_in[8];
  const float* b2  = (const float*)d_in[9];
  const float* g2  = (const float*)d_in[10];
  const float* be2 = (const float*)d_in[11];
  const float* W3  = (const float*)d_in[12];
  const float* b3  = (const float*)d_in[13];
  const float* g3  = (const float*)d_in[14];
  const float* be3 = (const float*)d_in[15];
  const float* Wc  = (const float*)d_in[16];
  const float* bc  = (const float*)d_in[17];
  const int*   y   = (const int*)d_in[18];
  const int*   order = (const int*)d_in[19];
  const int*   s1  = (const int*)d_in[20];
  const int*   s2  = (const int*)d_in[21];

  char* ws = (char*)d_ws;
  size_t off = 0;
  auto alloc = [&](size_t bytes) -> char* {
    char* p = ws + off;
    off += (bytes + 255) & ~(size_t)255;
    return p;
  };

  u16*   xs_bf   = (u16*)alloc((size_t)B_ROWS * DIM * 2);   // reused as hA after stage f
  u16*   feat_bf = (u16*)alloc((size_t)B_ROWS * DIM * 2);   // reused as hB after stage 2
  u16*   wf_bf   = (u16*)alloc((size_t)DIM * DIM * 2);
  u16*   w1_bf   = (u16*)alloc((size_t)DIM * DIM * 2);
  u16*   w2_bf   = (u16*)alloc((size_t)DIM * DIM * 2);
  u16*   w3_bf   = (u16*)alloc((size_t)DIM * DIM * 2);
  u16*   wc_bf   = (u16*)alloc((size_t)DIM * LDO * 2);
  float* bcp     = (float*)alloc(LDO * 4);
  float* Z       = (float*)alloc((size_t)B_ROWS * DIM * 4);
  float* proj    = (float*)alloc((size_t)B_ROWS * DIM * 4);
  float* outl    = (float*)alloc((size_t)B_ROWS * LDO * 4);
  float* cs      = (float*)alloc(DIM * 4);
  float* css     = (float*)alloc(DIM * 4);
  float* accs    = (float*)alloc(8 * 4);
  u16*   hA_bf   = xs_bf;
  u16*   hB_bf   = feat_bf;

  const int T = 256;
  const size_t wn = (size_t)DIM * DIM;
  const size_t an = (size_t)B_ROWS * DIM;

  // --- bf16 conversions ---
  k_conv_bf16<<<(unsigned)((wn / 4 + T - 1) / T), T, 0, stream>>>(Wf, wf_bf, wn);
  k_conv_bf16<<<(unsigned)((wn / 4 + T - 1) / T), T, 0, stream>>>(W1, w1_bf, wn);
  k_conv_bf16<<<(unsigned)((wn / 4 + T - 1) / T), T, 0, stream>>>(W2, w2_bf, wn);
  k_conv_bf16<<<(unsigned)((wn / 4 + T - 1) / T), T, 0, stream>>>(W3, w3_bf, wn);
  hipMemsetAsync(wc_bf, 0, (size_t)DIM * LDO * 2, stream);
  hipMemsetAsync(bcp, 0, LDO * 4, stream);
  {
    size_t pc = (size_t)DIM * NCLS;
    k_pad_wc<<<(unsigned)((pc + T - 1) / T), T, 0, stream>>>(Wc, bc, wc_bf, bcp);
  }
  k_gather_bf16<<<(unsigned)((an / 4 + T - 1) / T), T, 0, stream>>>(x, order, xs_bf);

  dim3 gN(DIM / BN, B_ROWS / BM);
  dim3 gC(LDO / BN, B_ROWS / BM);
  dim3 gst(DIM / T, 32);
  const unsigned gel = (unsigned)((an / 4 + T - 1) / T);

  // --- featurizer: Z = xs @ Wf + bf ; feat = relu(Z) ---
  k_gemm_bf16<<<gN, T, 0, stream>>>(xs_bf, DIM, wf_bf, DIM, bfv, Z, DIM, DIM);
  k_relu_bf16<<<gel, T, 0, stream>>>(Z, feat_bf, an);

  // --- classifier: out = feat @ Wc + bc (padded to 384 cols) ---
  k_gemm_bf16<<<gC, T, 0, stream>>>(feat_bf, DIM, wc_bf, LDO, bcp, outl, LDO, DIM);

  // --- layer 1 ---
  k_gemm_bf16<<<gN, T, 0, stream>>>(feat_bf, DIM, w1_bf, DIM, b1, Z, DIM, DIM);
  hipMemsetAsync(cs, 0, DIM * 4, stream);
  hipMemsetAsync(css, 0, DIM * 4, stream);
  k_colstats<<<gst, T, 0, stream>>>(Z, cs, css, B_ROWS / 32);
  k_bn_apply<<<gel, T, 0, stream>>>(Z, cs, css, g1, be1, hA_bf, nullptr, 0);

  // --- layer 2 ---
  k_gemm_bf16<<<gN, T, 0, stream>>>(hA_bf, DIM, w2_bf, DIM, b2, Z, DIM, DIM);
  hipMemsetAsync(cs, 0, DIM * 4, stream);
  hipMemsetAsync(css, 0, DIM * 4, stream);
  k_colstats<<<gst, T, 0, stream>>>(Z, cs, css, B_ROWS / 32);
  k_bn_apply<<<gel, T, 0, stream>>>(Z, cs, css, g2, be2, hB_bf, nullptr, 0);

  // --- layer 3 (projection, no relu, f32 out) ---
  k_gemm_bf16<<<gN, T, 0, stream>>>(hB_bf, DIM, w3_bf, DIM, b3, Z, DIM, DIM);
  hipMemsetAsync(cs, 0, DIM * 4, stream);
  hipMemsetAsync(css, 0, DIM * 4, stream);
  k_colstats<<<gst, T, 0, stream>>>(Z, cs, css, B_ROWS / 32);
  k_bn_apply<<<gel, T, 0, stream>>>(Z, cs, css, g3, be3, nullptr, proj, 1);

  // --- losses ---
  hipMemsetAsync(accs, 0, 8 * 4, stream);
  k_loss_logits<<<B_ROWS, T, 0, stream>>>(outl, s1, s2, lam, accs);
  k_loss_feat<<<B_ROWS, T, 0, stream>>>(proj, s1, s2, lam, accs);
  k_ce<<<B_ROWS, T, 0, stream>>>(outl, y, order, accs);
  k_final<<<1, 1, 0, stream>>>(accs, lam, (float*)d_out);
}
